// SlidingTileAttention2D_30477087932848
// MI455X (gfx1250) — compile-verified, matrix operand order checked by reading, not run on hardware
//
#include <hip/hip_runtime.h>

typedef __attribute__((ext_vector_type(16))) _Float16 v16h;
typedef __attribute__((ext_vector_type(8)))  _Float16 v8h;
typedef __attribute__((ext_vector_type(8)))  float    v8f;

#define SEQ     4096
#define HIDDEN  2048
#define NH      32
#define HD      64
#define NHD     2048   // NH*HD

// 1/sqrt(HD) * log2(e): folded into Wq so attention scores are base-2 ready.
#define QK_SCALE_LOG2E 0.18033688011112042f

// ---------------------------------------------------------------- WMMA core
__device__ __forceinline__ v8f wmma32(v16h a, v16h b, v8f c) {
  // D(16x16 f32) = A(16x32 f16) x B(32x16 f16) + C
  return __builtin_amdgcn_wmma_f32_16x16x32_f16(
      /*neg_a=*/false, a, /*neg_b=*/false, b,
      /*c_mod=*/(short)0, c, /*reuse_a=*/false, /*reuse_b=*/false);
}

// A/B fragment from LDS, contiguous-K layout.
// Per lane: row = lane%16; halves at [c .. c+7] and [c+16 .. c+23], c=(lane/16)*8.
__device__ __forceinline__ v16h load_frag(const _Float16* p) {
  v8h lo = *(const v8h*)p;
  v8h hi = *(const v8h*)(p + 16);
  v16h f;
#pragma unroll
  for (int i = 0; i < 8; ++i) { f[i] = lo[i]; f[i + 8] = hi[i]; }
  return f;
}

// ------------------------------------------------ CDNA5 async global->LDS
// Flat LDS aperture keeps workgroup-relative LDS byte offset in addr[31:0].
__device__ __forceinline__ unsigned lds_off(const void* p) {
  return (unsigned)(size_t)p;
}

// GVS mode: mem_addr = SGPR64 + VGPR32_offset; 16B per lane, tracked by ASYNCcnt.
__device__ __forceinline__ void async_copy16(const void* gbase, unsigned gbyteoff,
                                             const void* ldsdst) {
  asm volatile("global_load_async_to_lds_b128 %0, %1, %2"
               :
               : "v"(lds_off(ldsdst)), "v"(gbyteoff),
                 "s"((unsigned long long)(size_t)gbase)
               : "memory");
}
__device__ __forceinline__ void wait_async_le12() {
  asm volatile("s_wait_asynccnt 0xc" ::: "memory");
}
__device__ __forceinline__ void wait_async_le8() {
  asm volatile("s_wait_asynccnt 0x8" ::: "memory");
}
__device__ __forceinline__ void wait_async_le0() {
  asm volatile("s_wait_asynccnt 0x0" ::: "memory");
}

// ------------------------------------------------------- fp32 -> f16 (scaled)
__global__ void f32_to_f16_kernel(const float* __restrict__ src,
                                  _Float16* __restrict__ dst, float scale, int n) {
  int i = blockIdx.x * 256 + threadIdx.x;
  if (i < n) dst[i] = (_Float16)(src[i] * scale);
}

// ---------------------------------------------------------------- GEMM C=A*B^T
// A:[M,K] f16, B:[N,K] f16 (B^T used), C:[M,N].
// Block tile 256(M)x128(N), K-stage 64, double-buffered async staging.
// 8 waves: 4 (M) x 2 (N); wave tile 64x64 via 4x4 WMMA tiles.
template <typename OutT>
__global__ __launch_bounds__(256)
void gemm_wmma_kernel(const _Float16* __restrict__ A,
                      const _Float16* __restrict__ B,
                      OutT* __restrict__ C,
                      int M, int N, int K) {
  constexpr int LD = 72;  // padded row stride (halves): 144B, 16B aligned
  __shared__ __align__(16) _Float16 sA[2][256 * LD];
  __shared__ __align__(16) _Float16 sB[2][128 * LD];

  const int tid  = threadIdx.x;
  const int lane = tid & 31;
  const int wave = tid >> 5;
  const int wm   = wave & 3;       // M wave: 0..3 -> 64 rows each
  const int wn   = wave >> 2;      // N wave: 0..1 -> 64 cols each
  const int bm   = blockIdx.x * 256;
  const int bn   = blockIdx.y * 128;
  const int lr   = lane & 15;
  const int kh   = (lane >> 4) * 8;

  // Stage A(256x64) + B(128x64) halves: 2048 + 1024 16B chunks, 8+4 per thread
  // => 12 async instructions per wave per stage.
  auto stage = [&](int k0, int s) {
#pragma unroll
    for (int i = 0; i < 8; ++i) {
      int chunk = tid + i * 256;
      int row = chunk >> 3;       // 0..255
      int c8  = (chunk & 7) * 8;
      async_copy16(A, (unsigned)(((size_t)(bm + row) * K + k0 + c8) * 2),
                   &sA[s][row * LD + c8]);
    }
#pragma unroll
    for (int i = 0; i < 4; ++i) {
      int chunk = tid + i * 256;
      int row = chunk >> 3;       // 0..127
      int c8  = (chunk & 7) * 8;
      async_copy16(B, (unsigned)(((size_t)(bn + row) * K + k0 + c8) * 2),
                   &sB[s][row * LD + c8]);
    }
  };

  v8f acc[4][4] = {};

  stage(0, 0);
  int s = 0;
  for (int k0 = 0; k0 < K; k0 += 64, s ^= 1) {
    const bool more = (k0 + 64) < K;
    if (more) stage(k0 + 64, s ^ 1);
    if (more) wait_async_le12(); else wait_async_le0();
    __syncthreads();  // stage k0 visible to all waves

#pragma unroll
    for (int kk = 0; kk < 64; kk += 32) {
      v16h af[4];
      v16h bf[4];
#pragma unroll
      for (int mi = 0; mi < 4; ++mi)
        af[mi] = load_frag(&sA[s][(wm * 64 + mi * 16 + lr) * LD + kk + kh]);
#pragma unroll
      for (int ni = 0; ni < 4; ++ni)
        bf[ni] = load_frag(&sB[s][(wn * 64 + ni * 16 + lr) * LD + kk + kh]);
#pragma unroll
      for (int mi = 0; mi < 4; ++mi)
#pragma unroll
        for (int ni = 0; ni < 4; ++ni)
          acc[mi][ni] = wmma32(af[mi], bf[ni], acc[mi][ni]);
    }
    __syncthreads();  // all reads of buffer s done before it is re-staged
  }

  // C-frag layout: col = lane%16, VGPR r = row r + 8*(lane/16)
  const int chalf = lane >> 4;
  const int cn    = lane & 15;
#pragma unroll
  for (int mi = 0; mi < 4; ++mi)
#pragma unroll
    for (int ni = 0; ni < 4; ++ni)
#pragma unroll
      for (int r = 0; r < 8; ++r) {
        int row = bm + wm * 64 + mi * 16 + r + chalf * 8;
        int col = bn + wn * 64 + ni * 16 + cn;
        C[(size_t)row * N + col] = (OutT)acc[mi][ni][r];
      }
}

// ---------------------------------------------------------------- attention
// One workgroup per (query tile, head). 8 waves; wave owns 16 query rows.
// Flash-style base-2 online softmax over 9 neighbor KV tiles (Q pre-scaled by
// 0.125*log2e), double-buffered async KV staging, V transposed in LDS so all
// WMMA fragments load as contiguous ds_load_b128 pairs.
__global__ __launch_bounds__(256)
void sta_attn_kernel(const _Float16* __restrict__ Q,
                     const _Float16* __restrict__ K,
                     const _Float16* __restrict__ V,
                     _Float16* __restrict__ O) {
  constexpr int LD  = 72;
  constexpr int LDP = 136;  // P row stride (halves): 272B, 16B aligned
  constexpr int LDV = 136;  // Vt row stride (halves)
  __shared__ __align__(16) _Float16 sQ[128 * LD];
  __shared__ __align__(16) _Float16 sK[2][128 * LD];
  __shared__ __align__(16) _Float16 sV[2][128 * LD];
  __shared__ __align__(16) _Float16 sVt[64 * LDV];   // V^T: [hd][kv]
  __shared__ __align__(16) _Float16 sP[8 * 16 * LDP];

  const int t    = blockIdx.x;     // query tile 0..31
  const int h    = blockIdx.y;     // head 0..31
  const int tid  = threadIdx.x;
  const int lane = tid & 31;
  const int wave = tid >> 5;
  const int lr   = lane & 15;
  const int half = lane >> 4;
  const int kh   = half * 8;

  const int th = t >> 2, tw = t & 3;  // tile coords on 8x4 tile grid
  // STA clamped window center (KT=2 -> radius 1)
  const int ch = min(max(th, 1), 6);
  const int cw = min(max(tw, 1), 2);

  // KV tile staging: 4 K-chunks + 4 V-chunks per thread => 8 asyncs/wave/stage
  auto stage_kv = [&](int j, int s) {
    const int kt  = (ch + (j / 3) - 1) * 4 + (cw + (j % 3) - 1);
    const int kth = kt >> 2, ktw = kt & 3;
#pragma unroll
    for (int i = 0; i < 4; ++i) {
      int chunk = tid + i * 256;
      int rr = chunk >> 3;
      int c8 = (chunk & 7) * 8;
      int seq = (kth * 8 + (rr >> 4)) * 64 + ktw * 16 + (rr & 15);
      unsigned go = (unsigned)(((size_t)seq * NHD + h * HD + c8) * 2);
      async_copy16(K, go, &sK[s][rr * LD + c8]);
      async_copy16(V, go, &sV[s][rr * LD + c8]);
    }
  };

  // Prologue: async-stage Q tile (4 asyncs/wave, oldest) + KV stage 0.
#pragma unroll
  for (int i = 0; i < 4; ++i) {
    int chunk = tid + i * 256;
    int rr = chunk >> 3;
    int c8 = (chunk & 7) * 8;
    int seq = (th * 8 + (rr >> 4)) * 64 + tw * 16 + (rr & 15);
    async_copy16(Q, (unsigned)(((size_t)seq * NHD + h * HD + c8) * 2),
                 &sQ[rr * LD + c8]);
  }
  stage_kv(0, 0);

  float mrow[8], lrow[8];
#pragma unroll
  for (int r = 0; r < 8; ++r) { mrow[r] = -1e30f; lrow[r] = 0.0f; }
  v8f oacc[4] = {};
  v16h aq[2];                      // Q fragments, invariant over j

  _Float16* pw = &sP[wave * 16 * LDP];

  for (int j = 0; j < 9; ++j) {
    const int s = j & 1;
    const bool more = j < 8;
    if (more) stage_kv(j + 1, s ^ 1);     // target buffer free since barrier at end of j-1
    if (more) wait_async_le8(); else wait_async_le0();
    __syncthreads();  // Q (first iter) + KV stage j visible to all waves

    if (j == 0) {     // hoisted Q fragments (rows wave*16..wave*16+15)
#pragma unroll
      for (int kk = 0; kk < 2; ++kk)
        aq[kk] = load_frag(&sQ[(wave * 16 + lr) * LD + kk * 32 + kh]);
    }

    // Cooperative transpose: sV[s] (128 kv x 64 hd) -> sVt (64 hd x 128 kv)
#pragma unroll
    for (int i = 0; i < 4; ++i) {
      int chunk = tid + i * 256;
      int kv = chunk >> 3;
      int h8 = (chunk & 7) * 8;
      v8h d = *(const v8h*)&sV[s][kv * LD + h8];
#pragma unroll
      for (int e = 0; e < 8; ++e)
        sVt[(h8 + e) * LDV + kv] = d[e];
    }

    // S(16x128) = Qw(16x64) @ K^T(64x128); B^T rows of K^T are kv rows of K.
    // Scores arrive already multiplied by 0.125*log2e (folded into Wq).
    v8f sc[8] = {};
#pragma unroll
    for (int kk = 0; kk < 2; ++kk) {
#pragma unroll
      for (int n = 0; n < 8; ++n) {
        v16h bk = load_frag(&sK[s][(n * 16 + lr) * LD + kk * 32 + kh]);
        sc[n] = wmma32(aq[kk], bk, sc[n]);
      }
    }

    // rowwise max (16-lane half-wave reduction matches C-frag rows)
    float rmax[8];
#pragma unroll
    for (int r = 0; r < 8; ++r) rmax[r] = -1e30f;
#pragma unroll
    for (int n = 0; n < 8; ++n)
#pragma unroll
      for (int r = 0; r < 8; ++r) rmax[r] = fmaxf(rmax[r], sc[n][r]);
#pragma unroll
    for (int r = 0; r < 8; ++r) {
      rmax[r] = fmaxf(rmax[r], __shfl_xor(rmax[r], 1, 32));
      rmax[r] = fmaxf(rmax[r], __shfl_xor(rmax[r], 2, 32));
      rmax[r] = fmaxf(rmax[r], __shfl_xor(rmax[r], 4, 32));
      rmax[r] = fmaxf(rmax[r], __shfl_xor(rmax[r], 8, 32));
    }

    // base-2 online softmax
    float alpha[8], rsum[8];
#pragma unroll
    for (int r = 0; r < 8; ++r) {
      float mn = fmaxf(mrow[r], rmax[r]);
      alpha[r] = exp2f(mrow[r] - mn);
      mrow[r]  = mn;
      rsum[r]  = 0.0f;
    }
#pragma unroll
    for (int n = 0; n < 8; ++n)
#pragma unroll
      for (int r = 0; r < 8; ++r) {
        float p = exp2f(sc[n][r] - mrow[r]);
        sc[n][r] = p;
        rsum[r] += p;
      }
#pragma unroll
    for (int r = 0; r < 8; ++r) {
      rsum[r] += __shfl_xor(rsum[r], 1, 32);
      rsum[r] += __shfl_xor(rsum[r], 2, 32);
      rsum[r] += __shfl_xor(rsum[r], 4, 32);
      rsum[r] += __shfl_xor(rsum[r], 8, 32);
      lrow[r] = lrow[r] * alpha[r] + rsum[r];
    }
#pragma unroll
    for (int ni = 0; ni < 4; ++ni)
#pragma unroll
      for (int r = 0; r < 8; ++r) oacc[ni][r] *= alpha[r];

    // Re-layout P: C-frag -> row-major [16 x 128] f16 in LDS (per-wave region)
#pragma unroll
    for (int n = 0; n < 8; ++n)
#pragma unroll
      for (int r = 0; r < 8; ++r)
        pw[(r + 8 * half) * LDP + n * 16 + lr] = (_Float16)sc[n][r];
    __syncthreads();  // P and sVt visible before reload

    // O(16x64) += P(16x128) @ V(128x64); B^T rows now contiguous in sVt
#pragma unroll
    for (int kkk = 0; kkk < 4; ++kkk) {
      v16h ap = load_frag(&pw[lr * LDP + kkk * 32 + kh]);
#pragma unroll
      for (int ni = 0; ni < 4; ++ni) {
        v16h bv = load_frag(&sVt[(ni * 16 + lr) * LDV + kkk * 32 + kh]);
        oacc[ni] = wmma32(ap, bv, oacc[ni]);
      }
    }
    __syncthreads();  // reads of sK[s]/sV[s]/sVt done before restage/overwrite
  }

  // Normalize and scatter back to [SEQ, NH*HD] layout (f16)
#pragma unroll
  for (int r = 0; r < 8; ++r) {
    float inv = 1.0f / lrow[r];
    int qr  = wave * 16 + r + 8 * half;   // row within 128-token tile
    int seq = (th * 8 + (qr >> 4)) * 64 + tw * 16 + (qr & 15);
#pragma unroll
    for (int ni = 0; ni < 4; ++ni)
      O[(size_t)seq * NHD + h * HD + ni * 16 + lr] = (_Float16)(oacc[ni][r] * inv);
  }
}

// ---------------------------------------------------------------- launcher
extern "C" void kernel_launch(void* const* d_in, const int* in_sizes, int n_in,
                              void* d_out, int out_size, void* d_ws, size_t ws_size,
                              hipStream_t stream) {
  (void)in_sizes; (void)n_in; (void)out_size; (void)ws_size;

  const float* hs = (const float*)d_in[0];
  const float* Wq = (const float*)d_in[1];
  const float* Wk = (const float*)d_in[2];
  const float* Wv = (const float*)d_in[3];
  const float* Wo = (const float*)d_in[4];
  float* out = (float*)d_out;

  const int nHS = SEQ * HIDDEN;   // 8,388,608
  const int nW  = NHD * HIDDEN;   // 4,194,304

  _Float16* p  = (_Float16*)d_ws;
  _Float16* hX = p; p += nHS;     // hidden f16
  _Float16* wq = p; p += nW;
  _Float16* wk = p; p += nW;
  _Float16* wv = p; p += nW;
  _Float16* wo = p; p += nW;
  _Float16* q  = p; p += nHS;
  _Float16* k  = p; p += nHS;
  _Float16* v  = p; p += nHS;
  _Float16* o  = p; p += nHS;

  dim3 blk(256);
  f32_to_f16_kernel<<<nHS / 256, blk, 0, stream>>>(hs, hX, 1.0f, nHS);
  // Fold softmax scale (in base-2 units) into Wq so attention skips scaling.
  f32_to_f16_kernel<<<nW  / 256, blk, 0, stream>>>(Wq, wq, QK_SCALE_LOG2E, nW);
  f32_to_f16_kernel<<<nW  / 256, blk, 0, stream>>>(Wk, wk, 1.0f, nW);
  f32_to_f16_kernel<<<nW  / 256, blk, 0, stream>>>(Wv, wv, 1.0f, nW);
  f32_to_f16_kernel<<<nW  / 256, blk, 0, stream>>>(Wo, wo, 1.0f, nW);

  dim3 gq(SEQ / 256, NHD / 128);  // 16 x 16
  gemm_wmma_kernel<_Float16><<<gq, blk, 0, stream>>>(hX, wq, q, SEQ, NHD, HIDDEN);
  gemm_wmma_kernel<_Float16><<<gq, blk, 0, stream>>>(hX, wk, k, SEQ, NHD, HIDDEN);
  gemm_wmma_kernel<_Float16><<<gq, blk, 0, stream>>>(hX, wv, v, SEQ, NHD, HIDDEN);

  dim3 ga(32, 32);                // (tiles, heads)
  sta_attn_kernel<<<ga, blk, 0, stream>>>(q, k, v, o);

  dim3 go(SEQ / 256, HIDDEN / 128);
  gemm_wmma_kernel<float><<<go, blk, 0, stream>>>(o, wo, out, SEQ, HIDDEN, NHD);
}
